// GATLayer_33638183862456
// MI455X (gfx1250) — matrix operand order repaired and checked by reading, not compiled, not run on hardware
//
#include <hip/hip_runtime.h>
#include <hip/hip_bf16.h>

typedef __bf16 bf16_t;
typedef __attribute__((ext_vector_type(16))) __bf16 v16bf;
typedef __attribute__((ext_vector_type(8)))  float  v8f;
typedef __attribute__((ext_vector_type(4)))  float  v4f;
typedef __attribute__((ext_vector_type(4)))  int    v4i;

#define GAT_N     8192
#define GAT_F     128
#define ALPHA     0.2f
#define NEG_BIG   (-9.0e15f)

// Cross-half (lane ^ 16) exchange within wave32 via ds_swizzle SWAPX16
// (group-of-32 encoding: and=0x1f, or=0, xor=0x10 -> 0x401F)
__device__ __forceinline__ float swap16(float v) {
    int i = __builtin_amdgcn_ds_swizzle(__float_as_int(v), 0x401F);
    return __int_as_float(i);
}

// ---------------------------------------------------------------------------
// Kernel 1: Wh = h @ W (bf16 WMMA, f32 accumulate), store WhT (bf16, n-major),
//           exact f32 s1 = Wh@a1, s2 = Wh@a2 via LDS.
// grid: 512 blocks (16 rows each), 256 threads (8 waves, one 16x16 tile each)
// ---------------------------------------------------------------------------
__global__ __launch_bounds__(256) void k1_wh(
    const float* __restrict__ h, const float* __restrict__ W,
    const float* __restrict__ avec,
    bf16_t* __restrict__ WhT, float* __restrict__ s1, float* __restrict__ s2)
{
    __shared__ float WhS[16 * GAT_F];

    const int tid  = threadIdx.x;
    const int wave = tid >> 5;        // 0..7 -> output-feature group
    const int lane = tid & 31;
    const int half = lane >> 4;       // 0/1: which K sub-block this lane holds
    const int l16  = lane & 15;
    const int row0 = blockIdx.x * 16;
    const int row  = row0 + l16;      // A-matrix row (M)
    const int col  = wave * 16 + l16; // B/C column (N)

    v8f acc = {};
    for (int k0 = 0; k0 < GAT_F; k0 += 32) {
        // A: 16x32 bf16 tile of h. Lane layout: half==0 -> K = {0..7,16..23},
        //    half==1 -> K = {8..15,24..31} (relative to k0).
        const float* hp = h + (size_t)row * GAT_F + k0 + half * 8;
        v4f a0 = *(const v4f*)(hp);
        v4f a1 = *(const v4f*)(hp + 4);
        v4f a2 = *(const v4f*)(hp + 16);
        v4f a3 = *(const v4f*)(hp + 20);
        v16bf A;
#pragma unroll
        for (int i = 0; i < 4; ++i) {
            A[i]      = (bf16_t)a0[i];
            A[4 + i]  = (bf16_t)a1[i];
            A[8 + i]  = (bf16_t)a2[i];
            A[12 + i] = (bf16_t)a3[i];
        }
        // B: 32x16 bf16 tile of W. Lane holds column N=col, K rows
        //    half==0 -> k0+{0..7,16..23}, half==1 -> k0+{8..15,24..31}.
        const float* wp = W + (size_t)(k0 + half * 8) * GAT_F + col;
        v16bf B;
#pragma unroll
        for (int kk = 0; kk < 16; ++kk) B[kk] = (bf16_t)wp[(kk < 8 ? kk : kk + 8) * GAT_F];

        acc = __builtin_amdgcn_wmma_f32_16x16x32_bf16(
            false, A, false, B, (short)0, acc, false, false);
    }

    // C/D layout: vgpr v, lanes 0-15 -> M=v, lanes 16-31 -> M=v+8; N = l16.
#pragma unroll
    for (int v = 0; v < 8; ++v) {
        int r = v + half * 8;
        float val = acc[v];
        WhT[(size_t)col * GAT_N + (row0 + r)] = (bf16_t)val; // transposed bf16
        WhS[r * GAT_F + col] = val;                           // f32 for s1/s2
    }
    __syncthreads();

    if (tid < 16) {
        float d1 = 0.f, d2 = 0.f;
#pragma unroll 4
        for (int c = 0; c < GAT_F; ++c) {
            float wv = WhS[tid * GAT_F + c];
            d1 += wv * avec[c];
            d2 += wv * avec[GAT_F + c];
        }
        s1[row0 + tid] = d1;
        s2[row0 + tid] = d2;
    }
}

// ---------------------------------------------------------------------------
// Kernel 2: fused masked-softmax(leaky_relu(s1_i + s2_j)) @ Wh, flash style.
// grid: 512 blocks (16-row stripe), 128 threads (4 waves; wave w owns a
// 2048-column quarter). Online softmax per wave; merge partials via LDS.
// The accumulator rescale is skipped with a wave-uniform (scalar) branch
// whenever no row in the wave observed a new running max.
// ---------------------------------------------------------------------------
__global__ __launch_bounds__(128) void k2_attn(
    const int* __restrict__ adj, const bf16_t* __restrict__ WhT,
    const float* __restrict__ s1g, const float* __restrict__ s2g,
    float* __restrict__ out)
{
    __shared__ float O_lds[4][16][GAT_F];  // 32 KB
    __shared__ float m_arr[4][16];
    __shared__ float l_arr[4][16];

    const int tid   = threadIdx.x;
    const int wave  = tid >> 5;      // 0..3 column quarter
    const int lane  = tid & 31;
    const int half  = lane >> 4;
    const int l16   = lane & 15;
    const int row0  = blockIdx.x * 16;
    const int row   = row0 + l16;
    const int cbase = wave * (GAT_N / 4);

    const float s1v = s1g[row];
    const int* adjrow = adj + (size_t)row * GAT_N;

    float m = -3.0e38f;
    float l = 0.f;
    v8f acc[8];
#pragma unroll
    for (int g = 0; g < 8; ++g) acc[g] = (v8f){};

    for (int t = 0; t < (GAT_N / 4) / 32; ++t) {
        const int j0 = cbase + t * 32;
        const int kb = j0 + half * 16;  // lane's K cols: kb..kb+15

        v4i q0 = *(const v4i*)(adjrow + kb);
        v4i q1 = *(const v4i*)(adjrow + kb + 4);
        v4i q2 = *(const v4i*)(adjrow + kb + 8);
        v4i q3 = *(const v4i*)(adjrow + kb + 12);
        v4f z0 = *(const v4f*)(s2g + kb);
        v4f z1 = *(const v4f*)(s2g + kb + 4);
        v4f z2 = *(const v4f*)(s2g + kb + 8);
        v4f z3 = *(const v4f*)(s2g + kb + 12);

        float e[16];
#pragma unroll
        for (int i = 0; i < 4; ++i) {
            float x;
            // leaky_relu(x) = max(x, ALPHA*x) since 0 < ALPHA < 1
            x = s1v + z0[i]; x = fmaxf(x, ALPHA * x); e[i]      = (q0[i] > 0) ? x : NEG_BIG;
            x = s1v + z1[i]; x = fmaxf(x, ALPHA * x); e[4 + i]  = (q1[i] > 0) ? x : NEG_BIG;
            x = s1v + z2[i]; x = fmaxf(x, ALPHA * x); e[8 + i]  = (q2[i] > 0) ? x : NEG_BIG;
            x = s1v + z3[i]; x = fmaxf(x, ALPHA * x); e[12 + i] = (q3[i] > 0) ? x : NEG_BIG;
        }

        float tmax = e[0];
#pragma unroll
        for (int i = 1; i < 16; ++i) tmax = fmaxf(tmax, e[i]);
        tmax = fmaxf(tmax, swap16(tmax));   // row max across both lane halves

        // Wave-uniform skip: only rescale when some row got a new max.
        if (__builtin_amdgcn_ballot_w32(tmax > m) != 0u) {
            float mnew = fmaxf(m, tmax);
            float fs   = __expf(m - mnew);  // first iter: exp(-huge) = 0
            m = mnew;
            l *= fs;
#pragma unroll
            for (int g = 0; g < 8; ++g)
#pragma unroll
                for (int v = 0; v < 8; ++v) acc[g][v] *= fs;
        }

        v16bf A;
        float psum = 0.f;
#pragma unroll
        for (int i = 0; i < 16; ++i) {
            float p = __expf(e[i] - m);     // masked -> exp(-9e15 - m) = 0
            psum += p;
            A[i] = (bf16_t)p;
        }
        l += psum;

        const int jb = j0 + half * 16;      // B lane rows: jb..jb+15
#pragma unroll
        for (int g = 0; g < 8; ++g) {
            const int n = g * 16 + l16;
            v16bf B = *(const v16bf*)(WhT + (size_t)n * GAT_N + jb); // 32B contig
            acc[g] = __builtin_amdgcn_wmma_f32_16x16x32_bf16(
                false, A, false, B, (short)0, acc[g], false, false);
        }
    }

    // l: sum lane-half partials (both halves used identical scale factors)
    l += swap16(l);
    if (half == 0) { m_arr[wave][l16] = m; l_arr[wave][l16] = l; }
    __syncthreads();

    // rescale this wave's output to the global row max, dump to LDS
    float mf = fmaxf(fmaxf(m_arr[0][l16], m_arr[1][l16]),
                     fmaxf(m_arr[2][l16], m_arr[3][l16]));
    float fw = __expf(m - mf);
#pragma unroll
    for (int g = 0; g < 8; ++g)
#pragma unroll
        for (int v = 0; v < 8; ++v)
            O_lds[wave][v + half * 8][g * 16 + l16] = acc[g][v] * fw;
    __syncthreads();

    // merge 4 quarters, normalize, ELU, store (coalesced across threads)
    const int r  = tid >> 3;        // 0..15
    const int c0 = tid & 7;
    float mfr = fmaxf(fmaxf(m_arr[0][r], m_arr[1][r]),
                      fmaxf(m_arr[2][r], m_arr[3][r]));
    float lf = 0.f;
#pragma unroll
    for (int w = 0; w < 4; ++w) lf += l_arr[w][r] * __expf(m_arr[w][r] - mfr);
    const float inv = 1.0f / lf;

    float* op = out + (size_t)(row0 + r) * GAT_F;
#pragma unroll
    for (int i = 0; i < 16; ++i) {
        int c = c0 + 8 * i;
        float v = O_lds[0][r][c] + O_lds[1][r][c] + O_lds[2][r][c] + O_lds[3][r][c];
        v *= inv;
        op[c] = v > 0.f ? v : (__expf(v) - 1.0f);
    }
}

extern "C" void kernel_launch(void* const* d_in, const int* in_sizes, int n_in,
                              void* d_out, int out_size, void* d_ws, size_t ws_size,
                              hipStream_t stream) {
    (void)in_sizes; (void)n_in; (void)out_size; (void)ws_size;
    const float* h   = (const float*)d_in[0];
    const int*   adj = (const int*)  d_in[1];
    const float* W   = (const float*)d_in[2];
    const float* a   = (const float*)d_in[3];
    float* out = (float*)d_out;

    char* ws = (char*)d_ws;
    bf16_t* WhT = (bf16_t*)ws;                                   // 2 MB
    float*  s1  = (float*)(ws + (size_t)GAT_F * GAT_N * sizeof(bf16_t));
    float*  s2  = s1 + GAT_N;

    k1_wh  <<<GAT_N / 16, 256, 0, stream>>>(h, W, a, WhT, s1, s2);
    k2_attn<<<GAT_N / 16, 128, 0, stream>>>(adj, WhT, s1, s2, out);
}
